// GroupedQueryAttention_60206851555943
// MI455X (gfx1250) — compile-verified, matrix operand order checked by reading, not run on hardware
//
#include <hip/hip_runtime.h>
#include <hip/hip_bf16.h>

// ---------------- problem constants (from reference) ----------------
#define NB    2      // batch
#define SEQ   2048   // T
#define DMODEL 2048
#define NHEAD 16
#define NKV   4
#define HD    128    // head dim
#define NREP  (NHEAD / NKV)

typedef __attribute__((ext_vector_type(16))) __bf16 v16bf;
typedef __attribute__((ext_vector_type(8)))  float  v8f;

union FragU { v16bf v; uint4 u[2]; };

__device__ __forceinline__ unsigned short f2bf(float f) {
  unsigned int u = __float_as_uint(f);
  u += 0x7FFFu + ((u >> 16) & 1u);           // round-to-nearest-even
  return (unsigned short)(u >> 16);
}
__device__ __forceinline__ float bf2f(unsigned short h) {
  return __uint_as_float(((unsigned int)h) << 16);
}

__device__ __forceinline__ float redmax16(float x) {
  #pragma unroll
  for (int off = 1; off < 16; off <<= 1) x = fmaxf(x, __shfl_xor(x, off, 32));
  return x;
}
__device__ __forceinline__ float redsum16(float x) {
  #pragma unroll
  for (int off = 1; off < 16; off <<= 1) x += __shfl_xor(x, off, 32);
  return x;
}

// LDS byte-offset of a __shared__ pointer (generic -> addrspace(3) -> int)
typedef __attribute__((address_space(3))) unsigned short lds_ushort_t;
__device__ __forceinline__ unsigned lds_off(unsigned short* p) {
  return (unsigned)(unsigned long long)(lds_ushort_t*)p;
}

// gfx1250 async copy: 16B global -> LDS, tracked by ASYNCcnt (not LOADcnt).
__device__ __forceinline__ void async_b128(unsigned lds_addr, unsigned gl_off,
                                           const void* base) {
  asm volatile("global_load_async_to_lds_b128 %0, %1, %2"
               :: "v"(lds_addr), "v"(gl_off), "s"(base) : "memory");
}
__device__ __forceinline__ void wait_async3() {
  asm volatile("s_wait_asynccnt 0x3" ::: "memory");
}
__device__ __forceinline__ void wait_async0() {
  asm volatile("s_wait_asynccnt 0x0" ::: "memory");
}

// ---------------- fp32 -> bf16 conversion ----------------
__global__ void cvt_f32_bf16(const float* __restrict__ in,
                             unsigned short* __restrict__ out, int n) {
  int i = blockIdx.x * blockDim.x + threadIdx.x;
  int stride = gridDim.x * blockDim.x;
  for (; i < n; i += stride) out[i] = f2bf(in[i]);
}

// ---------------- bf16 WMMA GEMM: C[M,N] = A[M,K] * W[N,K]^T ----------------
// 256 threads = 8 waves (4 waveM x 2 waveN); block tile 64x128, K-step 32.
// Double-buffered LDS staging via GLOBAL_LOAD_ASYNC_TO_LDS_B128 (ASYNCcnt),
// rows padded to 80B (40 elems) for LDS bank spread; 16B aligned for b128.
#define AROW 40                       // padded row length (elements)
#define ATILE_B (64 * AROW * 2)       // 5120 B per A buffer
#define BTILE_B (128 * AROW * 2)      // 10240 B per B buffer
template <bool F32OUT>
__global__ void __launch_bounds__(256)
gemm_bf16(const unsigned short* __restrict__ A,
          const unsigned short* __restrict__ W,
          void* __restrict__ Cout, int M, int N, int K) {
  __shared__ unsigned short smem[(2 * ATILE_B + 2 * BTILE_B) / 2];
  const unsigned smem0 = lds_off(smem);

  const int lane = threadIdx.x & 31;
  const int wave = threadIdx.x >> 5;
  const int wm = wave & 3, wn = wave >> 2;
  const int half = lane >> 4;
  const int l16  = lane & 15;

  // issue one double-buffer stage: A tile 64x32 (256 x 16B), B tile 128x32 (512 x 16B)
  auto stage = [&](int kb, int b) {
    {
      const int row = threadIdx.x >> 2, c = threadIdx.x & 3;
      unsigned l = smem0 + (unsigned)(b * ATILE_B + row * (AROW * 2) + c * 16);
      unsigned g = (unsigned)((((size_t)(blockIdx.y * 64 + row)) * K + kb + c * 8) * 2);
      async_b128(l, g, A);
    }
    #pragma unroll
    for (int t = 0; t < 2; ++t) {
      const int idx = threadIdx.x + t * 256;
      const int row = idx >> 2, c = idx & 3;
      unsigned l = smem0 + (unsigned)(2 * ATILE_B + b * BTILE_B + row * (AROW * 2) + c * 16);
      unsigned g = (unsigned)((((size_t)(blockIdx.x * 128 + row)) * K + kb + c * 8) * 2);
      async_b128(l, g, W);
    }
  };

  v8f zero = {0.f,0.f,0.f,0.f,0.f,0.f,0.f,0.f};
  v8f acc[4];
  #pragma unroll
  for (int f = 0; f < 4; ++f) acc[f] = zero;

  stage(0, 0);
  int buf = 0;
  for (int kb = 0; kb < K; kb += 32) {
    if (kb + 32 < K) {
      stage(kb + 32, buf ^ 1);   // 3 new async ops in flight
      wait_async3();             // previous stage (3 ops) complete
    } else {
      wait_async0();
    }
    __syncthreads();             // stage `buf` visible to all waves

    const unsigned short* At = smem + (buf * ATILE_B) / 2;
    const unsigned short* Bt = smem + (2 * ATILE_B + buf * BTILE_B) / 2;
    FragU a;
    {
      const unsigned short* ar = At + (wm * 16 + l16) * AROW + half * 8;
      a.u[0] = *reinterpret_cast<const uint4*>(ar);
      a.u[1] = *reinterpret_cast<const uint4*>(ar + 16);
    }
    #pragma unroll
    for (int f = 0; f < 4; ++f) {
      const unsigned short* br = Bt + (wn * 64 + f * 16 + l16) * AROW + half * 8;
      FragU b;
      b.u[0] = *reinterpret_cast<const uint4*>(br);
      b.u[1] = *reinterpret_cast<const uint4*>(br + 16);
      acc[f] = __builtin_amdgcn_wmma_f32_16x16x32_bf16(
          false, a.v, false, b.v, (short)0, acc[f], false, false);
    }
    __syncthreads();             // everyone done reading `buf` before rewrite
    buf ^= 1;
  }

  #pragma unroll
  for (int f = 0; f < 4; ++f) {
    #pragma unroll
    for (int v = 0; v < 8; ++v) {
      const int m = blockIdx.y * 64 + wm * 16 + v + 8 * half;
      const int n = blockIdx.x * 128 + wn * 64 + f * 16 + l16;
      if (F32OUT) ((float*)Cout)[(size_t)m * N + n] = acc[f][v];
      else ((unsigned short*)Cout)[(size_t)m * N + n] = f2bf(acc[f][v]);
    }
  }
}

// ---------------- RoPE: (b,t,h,d) bf16 -> (b,h,t,d) bf16 ----------------
__global__ void rope_kernel(const unsigned short* __restrict__ src,
                            unsigned short* __restrict__ dst, int nheads) {
  const int total = NB * SEQ * nheads * (HD / 2);
  int idx = blockIdx.x * blockDim.x + threadIdx.x;
  if (idx >= total) return;
  const int p = idx % (HD / 2);
  const int h = (idx / (HD / 2)) % nheads;
  const int t = (idx / ((HD / 2) * nheads)) % SEQ;
  const int b = idx / ((HD / 2) * nheads * SEQ);
  const float freq = __powf(10000.f, -(float)(2 * p) / (float)HD);
  float s, c;
  __sincosf((float)t * freq, &s, &c);
  const size_t si = ((size_t)(b * SEQ + t) * nheads + h) * HD + 2 * p;
  const float x0 = bf2f(src[si]), x1 = bf2f(src[si + 1]);
  const size_t di = ((size_t)(b * nheads + h) * SEQ + t) * HD + 2 * p;
  dst[di]     = f2bf(x0 * c - x1 * s);
  dst[di + 1] = f2bf(x0 * s + x1 * c);
}

// ---------------- V transpose: (b,t,g,d) -> (b,g,d,t) ----------------
__global__ void vtrans_kernel(const unsigned short* __restrict__ src,
                              unsigned short* __restrict__ dst) {
  const int total = NB * SEQ * NKV * HD;
  int idx = blockIdx.x * blockDim.x + threadIdx.x;
  if (idx >= total) return;
  const int dd = idx % HD;
  const int g  = (idx / HD) % NKV;
  const int t  = (idx / (HD * NKV)) % SEQ;
  const int b  = idx / (HD * NKV * SEQ);
  dst[((size_t)(b * NKV + g) * HD + dd) * SEQ + t] =
      src[((size_t)(b * SEQ + t) * NKV + g) * HD + dd];
}

// ---------------- flash attention (causal, GQA) ----------------
// 128 threads = 4 waves per block; each wave owns one 16-row Q tile of one head.
// Per 32-key chunk: 8 WMMA for S = Q K^T, online softmax (16-lane shuffles),
// P staged via LDS (C-layout -> A-layout), 8 WMMA for O += P V.
__global__ void __launch_bounds__(128)
attn_kernel(const unsigned short* __restrict__ Qr,
            const unsigned short* __restrict__ Kr,
            const unsigned short* __restrict__ Vt,
            unsigned short* __restrict__ Oattn) {
  __shared__ unsigned short plds[4 * 16 * 32];
  const int lane = threadIdx.x & 31;
  const int wave = threadIdx.x >> 5;
  const int half = lane >> 4;
  const int l16  = lane & 15;

  const int wt = blockIdx.x * 4 + wave;   // global wave-tile id
  const int NT = SEQ / 16;                // 128 tiles per head
  const int tt = wt % NT;
  const int h  = (wt / NT) % NHEAD;
  const int b  = wt / (NT * NHEAD);
  const int g  = h / NREP;
  const int rowbase = tt * 16;

  const unsigned short* qbase = Qr + ((size_t)(b * NHEAD + h) * SEQ) * HD;
  const unsigned short* kbase = Kr + ((size_t)(b * NKV + g) * SEQ) * HD;
  const unsigned short* vbase = Vt + ((size_t)(b * NKV + g) * HD) * SEQ;

  // Q tile 16x128 as 4 A-fragments (16x32)
  FragU qf[4];
  {
    const unsigned short* qrow = qbase + (size_t)(rowbase + l16) * HD;
    #pragma unroll
    for (int f = 0; f < 4; ++f) {
      const int k0 = f * 32 + half * 8;
      qf[f].u[0] = *reinterpret_cast<const uint4*>(qrow + k0);
      qf[f].u[1] = *reinterpret_cast<const uint4*>(qrow + k0 + 16);
    }
  }

  v8f zero = {0.f,0.f,0.f,0.f,0.f,0.f,0.f,0.f};
  v8f o[8];
  #pragma unroll
  for (int i = 0; i < 8; ++i) o[i] = zero;
  float mrow[8], lrow[8];
  #pragma unroll
  for (int v = 0; v < 8; ++v) { mrow[v] = -1e30f; lrow[v] = 0.f; }

  const float scale = 0.088388347648318447f;  // 1/sqrt(128)
  unsigned short* myp = plds + wave * 16 * 32;
  const int limit = rowbase + 16;

  for (int jb = 0; jb < limit; jb += 32) {
    // ---- S = Q K^T (two 16x16 key tiles) ----
    v8f s1 = zero, s2 = zero;
    const unsigned short* krow1 = kbase + (size_t)(jb + l16) * HD;
    const unsigned short* krow2 = kbase + (size_t)(jb + 16 + l16) * HD;
    #pragma unroll
    for (int f = 0; f < 4; ++f) {
      const int k0 = f * 32 + half * 8;
      FragU b1, b2;
      b1.u[0] = *reinterpret_cast<const uint4*>(krow1 + k0);
      b1.u[1] = *reinterpret_cast<const uint4*>(krow1 + k0 + 16);
      b2.u[0] = *reinterpret_cast<const uint4*>(krow2 + k0);
      b2.u[1] = *reinterpret_cast<const uint4*>(krow2 + k0 + 16);
      s1 = __builtin_amdgcn_wmma_f32_16x16x32_bf16(false, qf[f].v, false, b1.v,
                                                   (short)0, s1, false, false);
      s2 = __builtin_amdgcn_wmma_f32_16x16x32_bf16(false, qf[f].v, false, b2.v,
                                                   (short)0, s2, false, false);
    }
    // ---- scale + causal mask + online softmax ----
    #pragma unroll
    for (int v = 0; v < 8; ++v) {
      const int m  = rowbase + v + 8 * half;
      const int j1 = jb + l16, j2 = jb + 16 + l16;
      const float a1 = (j1 <= m) ? s1[v] * scale : -1e30f;
      const float a2 = (j2 <= m) ? s2[v] * scale : -1e30f;
      const float mx = redmax16(fmaxf(a1, a2));
      const float mn = fmaxf(mrow[v], mx);
      const float corr = __expf(mrow[v] - mn);
      const float p1 = __expf(a1 - mn);
      const float p2 = __expf(a2 - mn);
      lrow[v] = lrow[v] * corr + redsum16(p1 + p2);
      mrow[v] = mn;
      #pragma unroll
      for (int nf = 0; nf < 8; ++nf) o[nf][v] = o[nf][v] * corr;
      const int r = (v + 8 * half) * 32;
      myp[r + l16]      = f2bf(p1);
      myp[r + 16 + l16] = f2bf(p2);
    }
    // ---- reload P as A-fragment (same wave; compiler inserts dscnt wait) ----
    FragU pa;
    {
      const unsigned short* prow = myp + l16 * 32;
      const int k0 = half * 8;
      pa.u[0] = *reinterpret_cast<const uint4*>(prow + k0);
      pa.u[1] = *reinterpret_cast<const uint4*>(prow + k0 + 16);
    }
    // ---- O += P V   (V^T layout: (d, T) rows contiguous in keys) ----
    #pragma unroll
    for (int nf = 0; nf < 8; ++nf) {
      const unsigned short* vrow = vbase + (size_t)(nf * 16 + l16) * SEQ + jb;
      FragU vb;
      const int k0 = half * 8;
      vb.u[0] = *reinterpret_cast<const uint4*>(vrow + k0);
      vb.u[1] = *reinterpret_cast<const uint4*>(vrow + k0 + 16);
      o[nf] = __builtin_amdgcn_wmma_f32_16x16x32_bf16(false, pa.v, false, vb.v,
                                                      (short)0, o[nf], false, false);
    }
  }

  // ---- epilogue: normalize, write (b,t,h*HD+n) bf16 for out-proj GEMM ----
  #pragma unroll
  for (int v = 0; v < 8; ++v) lrow[v] = 1.f / lrow[v];
  #pragma unroll
  for (int nf = 0; nf < 8; ++nf) {
    #pragma unroll
    for (int v = 0; v < 8; ++v) {
      const int t = rowbase + v + 8 * half;
      const int n = h * HD + nf * 16 + l16;
      Oattn[(size_t)(b * SEQ + t) * DMODEL + n] = f2bf(o[nf][v] * lrow[v]);
    }
  }
}

// ---------------- host orchestration ----------------
extern "C" void kernel_launch(void* const* d_in, const int* in_sizes, int n_in,
                              void* d_out, int out_size, void* d_ws, size_t ws_size,
                              hipStream_t stream) {
  const float* x  = (const float*)d_in[0];
  const float* wq = (const float*)d_in[1];
  const float* wk = (const float*)d_in[2];
  const float* wv = (const float*)d_in[3];
  const float* wo = (const float*)d_in[4];
  // d_in[5] = causal mask: applied analytically (j <= i) in attn_kernel.
  float* out = (float*)d_out;

  const size_t BT = (size_t)NB * SEQ;          // 4096 rows
  unsigned short* ws = (unsigned short*)d_ws;
  size_t off = 0;
  auto alloc = [&](size_t n) {
    unsigned short* p = ws + off;
    off += (n + 127) & ~(size_t)127;
    return p;
  };
  unsigned short* xb  = alloc(BT * DMODEL);          // x bf16
  unsigned short* wqb = alloc((size_t)DMODEL * DMODEL);
  unsigned short* wkb = alloc((size_t)NKV * HD * DMODEL);
  unsigned short* wvb = alloc((size_t)NKV * HD * DMODEL);
  unsigned short* wob = alloc((size_t)DMODEL * DMODEL);
  unsigned short* qb  = alloc(BT * DMODEL);          // (b,t,h,d) pre-rope
  unsigned short* kb  = alloc(BT * NKV * HD);
  unsigned short* vb  = alloc(BT * NKV * HD);
  unsigned short* Qr  = alloc(BT * DMODEL);          // (b,h,t,d)
  unsigned short* Kr  = alloc(BT * NKV * HD);        // (b,g,t,d)
  unsigned short* Vt  = alloc(BT * NKV * HD);        // (b,g,d,t)
  unsigned short* att = alloc(BT * DMODEL);          // (b,t,h,d)

  auto cvt = [&](const float* src, unsigned short* dst, size_t n) {
    int blocks = (int)((n + 255) / 256);
    if (blocks > 4096) blocks = 4096;
    cvt_f32_bf16<<<blocks, 256, 0, stream>>>(src, dst, (int)n);
  };
  cvt(x,  xb,  BT * DMODEL);
  cvt(wq, wqb, (size_t)DMODEL * DMODEL);
  cvt(wk, wkb, (size_t)NKV * HD * DMODEL);
  cvt(wv, wvb, (size_t)NKV * HD * DMODEL);
  cvt(wo, wob, (size_t)DMODEL * DMODEL);

  // QKV projections
  {
    dim3 g(DMODEL / 128, (unsigned)(BT / 64));
    gemm_bf16<false><<<g, 256, 0, stream>>>(xb, wqb, qb, (int)BT, DMODEL, DMODEL);
  }
  {
    dim3 g((NKV * HD) / 128, (unsigned)(BT / 64));
    gemm_bf16<false><<<g, 256, 0, stream>>>(xb, wkb, kb, (int)BT, NKV * HD, DMODEL);
    gemm_bf16<false><<<g, 256, 0, stream>>>(xb, wvb, vb, (int)BT, NKV * HD, DMODEL);
  }

  // RoPE + layouts
  {
    int nq = NB * SEQ * NHEAD * (HD / 2);
    rope_kernel<<<(nq + 255) / 256, 256, 0, stream>>>(qb, Qr, NHEAD);
    int nk = NB * SEQ * NKV * (HD / 2);
    rope_kernel<<<(nk + 255) / 256, 256, 0, stream>>>(kb, Kr, NKV);
    int nv = NB * SEQ * NKV * HD;
    vtrans_kernel<<<(nv + 255) / 256, 256, 0, stream>>>(vb, Vt);
  }

  // Flash attention: NB*NHEAD*(SEQ/16) wave-tiles, 4 waves/block
  {
    int tiles = NB * NHEAD * (SEQ / 16);   // 4096
    attn_kernel<<<tiles / 4, 128, 0, stream>>>(Qr, Kr, Vt, att);
  }

  // Output projection -> fp32 d_out
  {
    dim3 g(DMODEL / 128, (unsigned)(BT / 64));
    gemm_bf16<true><<<g, 256, 0, stream>>>(att, wob, out, (int)BT, DMODEL, DMODEL);
  }
  (void)in_sizes; (void)n_in; (void)out_size; (void)ws_size;
}
